// MLPList_58574763983287
// MI455X (gfx1250) — matrix operand order repaired and checked by reading, not compiled, not run on hardware
//
#include <hip/hip_runtime.h>

typedef __attribute__((ext_vector_type(16))) _Float16 v16h;
typedef __attribute__((ext_vector_type(8)))  float    v8f;

#define BATCH     32768
#define DIN       64
#define HID       64
#define NEXP      63
#define TILE_B    128
#define NTHREADS  256

// ---- fragment loaders for 16-bit WMMA (contiguous k order, same for A and B) ----

// A (16x32, MxK): lane holds row M=lane&15. halves 0..15 -> K = kb+0..15,
// kb = 16*(lane>=16). Caller folds kc*32 into base.
__device__ __forceinline__ v16h load_a_frag(const _Float16* __restrict__ row_base,
                                            int hsel) {
  const _Float16* p = row_base + hsel * 16;
  v16h f;
#pragma unroll
  for (int h = 0; h < 8; ++h) f[h] = p[h];
#pragma unroll
  for (int h = 0; h < 8; ++h) f[8 + h] = p[8 + h];
  return f;
}

// B (32x16, KxN): lane holds column N=lane&15; halves 0..15 -> K = 16*(lane>=16)+h.
// Wt is stored transposed in LDS as [n][k], so the 16 halves are contiguous.
__device__ __forceinline__ v16h load_b_frag(const _Float16* __restrict__ col_base,
                                            int hsel) {
  const _Float16* p = col_base + hsel * 16;
  v16h f;
#pragma unroll
  for (int h = 0; h < 16; ++h) f[h] = p[h];
  return f;
}

// One dense layer for this wave's 16 rows:
//   Aout[row0..row0+15][0..63] = relu(Ain[row0..][0..63] @ Wt^T + bias)
__device__ __forceinline__ void layer_gemm(const _Float16* __restrict__ Ain,
                                           const _Float16* __restrict__ Wt,
                                           const float* __restrict__ bias,
                                           _Float16* __restrict__ Aout,
                                           int row0, int lrow, int hsel) {
  const v8f zero = {0.f, 0.f, 0.f, 0.f, 0.f, 0.f, 0.f, 0.f};
  v8f acc[4] = {zero, zero, zero, zero};
#pragma unroll
  for (int kc = 0; kc < 2; ++kc) {
    const v16h a = load_a_frag(&Ain[(row0 + lrow) * 64 + kc * 32], hsel);
#pragma unroll
    for (int nt = 0; nt < 4; ++nt) {
      const v16h b = load_b_frag(&Wt[(nt * 16 + lrow) * 64 + kc * 32], hsel);
      acc[nt] = __builtin_amdgcn_wmma_f32_16x16x32_f16(
          false, a, false, b, (short)0, acc[nt], false, false);
    }
  }
  // C/D layout: element (M = v + 8*hsel, N = lane&15) in VGPR v
#pragma unroll
  for (int nt = 0; nt < 4; ++nt) {
    const int n = nt * 16 + lrow;
    const float bb = bias[n];
#pragma unroll
    for (int v = 0; v < 8; ++v) {
      const int m = row0 + hsel * 8 + v;
      float val = acc[nt][v] + bb;
      val = fmaxf(val, 0.0f);
      Aout[m * 64 + n] = (_Float16)val;
    }
  }
}

__global__ __launch_bounds__(NTHREADS)
void mlp_experts(const float* __restrict__ X,
                 const float* __restrict__ W1, const float* __restrict__ b1,
                 const float* __restrict__ W2, const float* __restrict__ b2,
                 const float* __restrict__ W3, const float* __restrict__ b3,
                 float* __restrict__ out) {
  __shared__ _Float16 sW1t[HID * DIN];     // transposed [n][k], causally masked
  __shared__ _Float16 sW2t[HID * HID];     // transposed [n][k]
  __shared__ _Float16 sW3t[16 * HID];      // [n][k]; row n==0 = w3, rest zero
  __shared__ float    sB1[HID];
  __shared__ float    sB2[HID];
  __shared__ _Float16 sX[TILE_B * DIN];    // activation ping (X, then h2)
  __shared__ _Float16 sH[TILE_B * HID];    // activation pong (h1)

  const int r    = blockIdx.x;             // expert r -> output dim r+1
  const int tile = blockIdx.y;
  const int tid  = threadIdx.x;
  const int lane = tid & 31;
  const int wave = tid >> 5;
  const int lrow = lane & 15;
  const int hsel = lane >> 4;
  const float b3r = b3[r];

  // ---- stage weights (transpose to LDS f16; mask W1 rows j > r) ----
  const float* W1r = W1 + (size_t)r * DIN * HID;
  const float* W2r = W2 + (size_t)r * HID * HID;
  for (int e = tid; e < DIN * HID; e += NTHREADS) {
    const int j = e >> 6;                  // input index k
    const int h = e & 63;                  // output index n
    const float w1 = (j <= r) ? W1r[e] : 0.0f;
    sW1t[h * DIN + j] = (_Float16)w1;
    sW2t[h * HID + j] = (_Float16)W2r[e];
  }
  // w3 as a zero-padded 64x16 B operand: column 0 = w3, columns 1..15 = 0
  for (int e = tid; e < 16 * HID; e += NTHREADS) {
    const int n = e >> 6;
    const int k = e & 63;
    sW3t[e] = (n == 0) ? (_Float16)W3[r * HID + k] : (_Float16)0.0f;
  }
  if (tid < HID) {
    sB1[tid] = b1[r * HID + tid];
    sB2[tid] = b2[r * HID + tid];
  }
  // ---- stage input tile as f16 (vectorized f32x4 global reads) ----
  const float4* Xt = reinterpret_cast<const float4*>(X + (size_t)tile * TILE_B * DIN);
  for (int e = tid; e < TILE_B * DIN / 4; e += NTHREADS) {
    const float4 v = Xt[e];
    _Float16* p = &sX[e * 4];
    p[0] = (_Float16)v.x; p[1] = (_Float16)v.y;
    p[2] = (_Float16)v.z; p[3] = (_Float16)v.w;
  }
  __syncthreads();

  const int row0 = wave * 16;              // this wave's private 16 rows

  // layer 1: sH = relu(sX @ W1m + b1); layer 2: sX = relu(sH @ W2 + b2)
  // (per-wave row ownership + in-order per-wave LDS => no extra barriers)
  layer_gemm(sX, sW1t, sB1, sH, row0, lrow, hsel);
  layer_gemm(sH, sW2t, sB2, sX, row0, lrow, hsel);

  // layer 3 on WMMA: o = h2 @ [w3 | 0...] ; column 0 of D holds the result.
  {
    const v8f zero = {0.f, 0.f, 0.f, 0.f, 0.f, 0.f, 0.f, 0.f};
    v8f acc = zero;
#pragma unroll
    for (int kc = 0; kc < 2; ++kc) {
      const v16h a = load_a_frag(&sX[(row0 + lrow) * 64 + kc * 32], hsel);
      const v16h b = load_b_frag(&sW3t[lrow * 64 + kc * 32], hsel);
      acc = __builtin_amdgcn_wmma_f32_16x16x32_f16(
          false, a, false, b, (short)0, acc, false, false);
    }
    // lanes with N==0 (lanes 0 and 16) hold the output column:
    // VGPR v -> batch row row0 + 8*hsel + v
    if (lrow == 0) {
      const size_t bbase = (size_t)tile * TILE_B + row0 + hsel * 8;
#pragma unroll
      for (int v = 0; v < 8; ++v) {
        out[(bbase + v) * 64 + (r + 1)] = acc[v] + b3r;
      }
    }
  }
}

// Dim 0: constant MLP on 0.1*linspace(-5,5,10); broadcast into out[:, 0].
__global__ __launch_bounds__(256)
void mlp_dim0(const float* __restrict__ w0_1, const float* __restrict__ b0_1,
              const float* __restrict__ w0_2, const float* __restrict__ b0_2,
              const float* __restrict__ w0_3, const float* __restrict__ b0_3,
              float* __restrict__ out) {
  __shared__ float h1s[64];
  __shared__ float h2s[64];
  __shared__ float o0s;
  const int tid = threadIdx.x;
  if (tid < 64) {
    float s = b0_1[tid];
#pragma unroll
    for (int k = 0; k < 10; ++k) {
      const float v = 0.1f * (-5.0f + 10.0f * (float)k / 9.0f);
      s += v * w0_1[k * 64 + tid];
    }
    h1s[tid] = fmaxf(s, 0.0f);
  }
  __syncthreads();
  if (tid < 64) {
    float s = b0_2[tid];
    for (int k = 0; k < 64; ++k) s += h1s[k] * w0_2[k * 64 + tid];
    h2s[tid] = fmaxf(s, 0.0f);
  }
  __syncthreads();
  if (tid == 0) {
    float s = b0_3[0];
    for (int k = 0; k < 64; ++k) s += h2s[k] * w0_3[k];
    o0s = s;
  }
  __syncthreads();
  out[((size_t)blockIdx.x * 256 + tid) * 64] = o0s;
}

extern "C" void kernel_launch(void* const* d_in, const int* in_sizes, int n_in,
                              void* d_out, int out_size, void* d_ws, size_t ws_size,
                              hipStream_t stream) {
  const float* X    = (const float*)d_in[0];
  const float* W1   = (const float*)d_in[1];
  const float* b1   = (const float*)d_in[2];
  const float* W2   = (const float*)d_in[3];
  const float* b2   = (const float*)d_in[4];
  const float* W3   = (const float*)d_in[5];
  const float* b3   = (const float*)d_in[6];
  const float* w0_1 = (const float*)d_in[7];
  const float* b0_1 = (const float*)d_in[8];
  const float* w0_2 = (const float*)d_in[9];
  const float* b0_2 = (const float*)d_in[10];
  const float* w0_3 = (const float*)d_in[11];
  const float* b0_3 = (const float*)d_in[12];
  float* out = (float*)d_out;

  dim3 grid(NEXP, BATCH / TILE_B);
  mlp_experts<<<grid, NTHREADS, 0, stream>>>(X, W1, b1, W2, b2, W3, b3, out);
  mlp_dim0<<<BATCH / 256, 256, 0, stream>>>(w0_1, b0_1, w0_2, b0_2, w0_3, b0_3, out);
}
